// ForgetAttention_39702677684761
// MI455X (gfx1250) — matrix operand order repaired and checked by reading, not compiled, not run on hardware
//
#include <hip/hip_runtime.h>
#include <stdint.h>

// ---------------------------------------------------------------------------
// ForgetAttention for MI455X (gfx1250, wave32, WMMA).
// Chunked linear-attention formulation: all heavy math is
// v_wmma_f32_16x16x32_bf16 with fp32 accumulation.
// ---------------------------------------------------------------------------

#define S_LEN 2048
#define BATCH 4
#define HID   1024
#define NH    16
#define HD    64
#define CHK   64
#define NCHK  (S_LEN / CHK)   // 32
#define BH    (BATCH * NH)    // 64
#define MROWS (S_LEN * BATCH) // 8192

typedef unsigned short u16t;
typedef __attribute__((ext_vector_type(16))) __bf16 v16bf;
typedef __attribute__((ext_vector_type(8)))  float  v8f;

static __device__ __forceinline__ u16t f2bf(float x) {
  union { float f; unsigned u; } c; c.f = x;
  unsigned u = c.u;
  unsigned r = (u + 0x7FFFu + ((u >> 16) & 1u)) >> 16;
  return (u16t)r;
}
static __device__ __forceinline__ float bf2f(u16t v) {
  union { unsigned u; float f; } c; c.u = ((unsigned)v) << 16; return c.f;
}

// A fragment, 16x32 bf16 (ISA 7.12.2): lane lr=lane&15 holds row m0+lr,
// K chunks [kh*8, kh*8+8) and [16+kh*8, 16+kh*8+8), kh = lane>>4.
static __device__ __forceinline__ v16bf load_a_frag(const u16t* base, int ld,
                                                    int r0, int kk, int lane) {
  int lr = lane & 15, kh = (lane >> 4) & 1;
  const u16t* p = base + (size_t)(r0 + lr) * ld + kk + kh * 8;
  v16bf f;
  ((uint4*)&f)[0] = *(const uint4*)p;
  ((uint4*)&f)[1] = *(const uint4*)(p + 16);
  return f;
}
// B fragment, 32x16 bf16: lane lr holds column n0+lr, K chunks
// [kh*8, kh*8+8) and [16+kh*8, 16+kh*8+8). Source stored row-major as [n][k].
static __device__ __forceinline__ v16bf load_b_frag(const u16t* base, int ld,
                                                    int n0, int kk, int lane) {
  int lr = lane & 15, kh = (lane >> 4) & 1;
  const u16t* p = base + (size_t)(n0 + lr) * ld + kk + kh * 8;
  v16bf f;
  ((uint4*)&f)[0] = *(const uint4*)p;
  ((uint4*)&f)[1] = *(const uint4*)(p + 16);
  return f;
}

// ---------------------------------------------------------------------------
__global__ __launch_bounds__(256) void convert_f32_bf16(const float* __restrict__ src,
                                                        u16t* __restrict__ dst, int n) {
  for (int i = blockIdx.x * blockDim.x + threadIdx.x; i < n; i += gridDim.x * blockDim.x)
    dst[i] = f2bf(src[i]);
}

__global__ __launch_bounds__(256) void fill_zero_f32(float* __restrict__ dst, int n) {
  for (int i = blockIdx.x * blockDim.x + threadIdx.x; i < n; i += gridDim.x * blockDim.x)
    dst[i] = 0.0f;
}

// ---------------------------------------------------------------------------
// Generic bf16 GEMM: out[m][n] = sum_k A[m][k] * W[n][k] + bias[n].
// Each wave computes a 64 x (16*NSUB) block: 4 M-tiles x NSUB N-tiles,
// so per k-step: (4 + NSUB) fragment loads feed 4*NSUB WMMAs.
// mode 0: phi = elu(x)+1, store bf16   (Q/K projections)
// mode 1: plain, store bf16            (V projection)
// mode 2: plain, store f32             (final output projection)
// mode 3: sigmoid, store f32           (forget gate)
template <int NSUB>
__global__ __launch_bounds__(256) void gemm_bf16_kernel(
    const u16t* __restrict__ A, const u16t* __restrict__ Bw,
    const float* __restrict__ bias, void* __restrict__ outp,
    int Mdim, int Ndim, int Kdim, int mode) {
  int lane = threadIdx.x & 31;
  int wave = blockIdx.x * (blockDim.x >> 5) + (threadIdx.x >> 5);
  int ntiles = Ndim / (16 * NSUB);
  int n0 = (wave % ntiles) * (16 * NSUB);
  int m0 = (wave / ntiles) << 6;
  if (m0 >= Mdim) return;

  v8f acc[4][NSUB];
#pragma unroll
  for (int t = 0; t < 4; t++)
#pragma unroll
    for (int s = 0; s < NSUB; s++) acc[t][s] = (v8f){0, 0, 0, 0, 0, 0, 0, 0};

  int lr = lane & 15;
  for (int kk = 0; kk < Kdim; kk += 32) {
    if (kk + 32 < Kdim) {
      __builtin_prefetch(Bw + (size_t)(n0 + lr) * Kdim + kk + 32, 0, 3);
      __builtin_prefetch(A + (size_t)(m0 + lr) * Kdim + kk + 32, 0, 3);
    }
    v16bf bfrag[NSUB];
#pragma unroll
    for (int s = 0; s < NSUB; s++)
      bfrag[s] = load_b_frag(Bw, Kdim, n0 + s * 16, kk, lane);
#pragma unroll
    for (int t = 0; t < 4; t++) {
      v16bf afrag = load_a_frag(A, Kdim, m0 + t * 16, kk, lane);
#pragma unroll
      for (int s = 0; s < NSUB; s++)
        acc[t][s] = __builtin_amdgcn_wmma_f32_16x16x32_bf16(
            false, afrag, false, bfrag[s], (short)0, acc[t][s], false, false);
    }
  }

  int half = lane >> 4;
#pragma unroll
  for (int s = 0; s < NSUB; s++) {
    int col = n0 + s * 16 + lr;
    float bv = bias[col];
#pragma unroll
    for (int t = 0; t < 4; t++) {
#pragma unroll
      for (int r = 0; r < 8; r++) {
        int row = m0 + t * 16 + half * 8 + r;
        float val = acc[t][s][r] + bv;
        size_t oi = (size_t)row * Ndim + col;
        if (mode == 0) {
          float ph = (val >= 0.0f) ? (val + 1.0f) : __expf(val);  // elu(x)+1
          ((u16t*)outp)[oi] = f2bf(ph);
        } else if (mode == 1) {
          ((u16t*)outp)[oi] = f2bf(val);
        } else if (mode == 2) {
          ((float*)outp)[oi] = val;
        } else {
          ((float*)outp)[oi] = 1.0f / (1.0f + __expf(-val));
        }
      }
    }
  }
}

// ---------------------------------------------------------------------------
// Per-chunk local reduction: U = K''^T V, zu = rowsum(K''^T), aC = prod(f).
// K''_j = (aC/a_j) * phi_k_j.   One block (4 waves) per (chunk, b, h).
__global__ __launch_bounds__(128) void chunk_local_kernel(
    const u16t* __restrict__ phik, const u16t* __restrict__ vbuf,
    const float* __restrict__ fg, float* __restrict__ Ubuf,
    float* __restrict__ zubuf, float* __restrict__ aCbuf) {
  __shared__ float aArr[CHK];
  __shared__ __attribute__((aligned(16))) u16t KsuT[HD * CHK];  // [d][j]
  __shared__ __attribute__((aligned(16))) u16t VbT[HD * CHK];   // [e][j]
  int job = blockIdx.x;
  int bh = job & 63, c = job >> 6;
  int b = bh >> 4, h = bh & 15;
  int tid = threadIdx.x;

  if (tid == 0) {
    float a = 1.0f;
    for (int t = 0; t < CHK; t++) {
      int m = (c * CHK + t) * BATCH + b;
      a *= fg[(size_t)m * NH + h];
      if (a < 1e-30f) a = 1e-30f;
      aArr[t] = a;
    }
  }
  __syncthreads();
  float aC = aArr[CHK - 1];

  for (int idx = tid; idx < HD * CHK; idx += 128) {
    int j = idx >> 6, x = idx & 63;
    int m = (c * CHK + j) * BATCH + b;
    size_t base = (size_t)m * HID + h * HD;
    KsuT[x * CHK + j] = f2bf(bf2f(phik[base + x]) * (aC / aArr[j]));
    VbT[x * CHK + j] = f2bf(bf2f(vbuf[base + x]));
  }
  __syncthreads();

  int lane = tid & 31, wv = tid >> 5;
  int lr = lane & 15, half = lane >> 4;
  float* Uout = Ubuf + (size_t)job * (HD * HD);
#pragma unroll
  for (int i = 0; i < 4; i++) {
    int tt = wv * 4 + i;
    int d0 = (tt >> 2) << 4, e0 = (tt & 3) << 4;
    v8f acc = (v8f){0, 0, 0, 0, 0, 0, 0, 0};
#pragma unroll
    for (int kk = 0; kk < CHK; kk += 32) {
      v16bf af = load_a_frag(KsuT, CHK, d0, kk, lane);
      v16bf bfr = load_b_frag(VbT, CHK, e0, kk, lane);
      acc = __builtin_amdgcn_wmma_f32_16x16x32_bf16(false, af, false, bfr,
                                                    (short)0, acc, false, false);
    }
#pragma unroll
    for (int r = 0; r < 8; r++)
      Uout[(d0 + half * 8 + r) * HD + e0 + lr] = acc[r];
  }

  if (tid < HD) {
    float s = 0.0f;
    for (int j = 0; j < CHK; j++) s += bf2f(KsuT[tid * CHK + j]);
    zubuf[(size_t)job * HD + tid] = s;
  }
  if (tid == 0) aCbuf[job] = aC;
}

// ---------------------------------------------------------------------------
// Cross-chunk scan (only 32 sequential steps), states in registers.
// Checkpoints are the state ENTERING each chunk.
__global__ __launch_bounds__(256) void state_scan_kernel(
    const float* __restrict__ Ubuf, const float* __restrict__ zubuf,
    const float* __restrict__ aCbuf, float* __restrict__ Sck,
    float* __restrict__ zck) {
  int bh = blockIdx.x;
  int tid = threadIdx.x;
  float s[16];
#pragma unroll
  for (int i = 0; i < 16; i++) s[i] = 0.0f;
  float zv = 0.0f;
  for (int c = 0; c < NCHK; c++) {
    size_t job = (size_t)c * BH + bh;
    float* so = Sck + job * (HD * HD) + tid * 16;
#pragma unroll
    for (int i = 0; i < 16; i++) so[i] = s[i];
    if (tid < HD) zck[job * HD + tid] = zv;
    float g = aCbuf[job];
    const float* uu = Ubuf + job * (HD * HD) + tid * 16;
#pragma unroll
    for (int i = 0; i < 16; i++) s[i] = g * s[i] + uu[i];
    if (tid < HD) zv = g * zv + zubuf[job * HD + tid];
  }
}

// ---------------------------------------------------------------------------
// Per-chunk output:
//   P[t][j] = (a_t/a_j)(phi_q_t . phi_k_j), causal-masked      (WMMA GEMM)
//   num     = P * V + (a_t phi_q_t) * S_start                  (WMMA GEMMs)
//   den_t   = a_t (phi_q_t . z_start) + rowsum(P[t]) + 1e-6
//   AO[t][e] = num/den  (bf16, fed to final Wo GEMM)
__global__ __launch_bounds__(128) void chunk_out_kernel(
    const u16t* __restrict__ phiq, const u16t* __restrict__ phik,
    const u16t* __restrict__ vbuf, const float* __restrict__ fg,
    const float* __restrict__ Sck, const float* __restrict__ zck,
    u16t* __restrict__ AO) {
  __shared__ float aArr[CHK];
  __shared__ float rowsum[CHK];
  __shared__ float pqz[CHK];
  __shared__ __attribute__((aligned(16))) u16t Qs[CHK * HD];   // [t][d]
  __shared__ __attribute__((aligned(16))) u16t Kq[CHK * HD];   // [j][d]
  __shared__ __attribute__((aligned(16))) u16t VbT[HD * CHK];  // [e][j]
  __shared__ __attribute__((aligned(16))) u16t SsT[HD * HD];   // [e][d]
  __shared__ __attribute__((aligned(16))) u16t Pm[CHK * CHK];  // [t][j]
  int job = blockIdx.x;
  int bh = job & 63, c = job >> 6;
  int b = bh >> 4, h = bh & 15;
  int tid = threadIdx.x;

  if (tid == 0) {
    float a = 1.0f;
    for (int t = 0; t < CHK; t++) {
      int m = (c * CHK + t) * BATCH + b;
      a *= fg[(size_t)m * NH + h];
      if (a < 1e-30f) a = 1e-30f;
      aArr[t] = a;
    }
  }
  if (tid < CHK) rowsum[tid] = 0.0f;
  __syncthreads();

  const float* zs = zck + (size_t)job * HD;
  for (int idx = tid; idx < HD * CHK; idx += 128) {
    int r = idx >> 6, x = idx & 63;
    int m = (c * CHK + r) * BATCH + b;
    size_t base = (size_t)m * HID + h * HD;
    float pq = bf2f(phiq[base + x]);
    float pk = bf2f(phik[base + x]);
    Qs[r * HD + x] = f2bf(aArr[r] * pq);
    Kq[r * HD + x] = f2bf(pk / aArr[r]);
    VbT[x * CHK + r] = f2bf(bf2f(vbuf[base + x]));
    SsT[x * HD + r] = f2bf(Sck[(size_t)job * (HD * HD) + r * HD + x]);
  }
  if (tid < CHK) {
    int m = (c * CHK + tid) * BATCH + b;
    size_t base = (size_t)m * HID + h * HD;
    float s = 0.0f;
    for (int d = 0; d < HD; d++) s += bf2f(phiq[base + d]) * zs[d];
    pqz[tid] = s;
  }
  __syncthreads();

  int lane = tid & 31, wv = tid >> 5;
  int lr = lane & 15, half = lane >> 4;

  // P = Qs x Kq^T  (causal mask + row sums into LDS)
#pragma unroll
  for (int i = 0; i < 4; i++) {
    int tt = wv * 4 + i;
    int t0 = (tt >> 2) << 4, j0 = (tt & 3) << 4;
    v8f acc = (v8f){0, 0, 0, 0, 0, 0, 0, 0};
#pragma unroll
    for (int kk = 0; kk < HD; kk += 32) {
      v16bf af = load_a_frag(Qs, HD, t0, kk, lane);
      v16bf bfr = load_b_frag(Kq, HD, j0, kk, lane);
      acc = __builtin_amdgcn_wmma_f32_16x16x32_bf16(false, af, false, bfr,
                                                    (short)0, acc, false, false);
    }
#pragma unroll
    for (int r = 0; r < 8; r++) {
      int tr = t0 + half * 8 + r;
      int jc = j0 + lr;
      float val = (jc <= tr) ? acc[r] : 0.0f;
      Pm[tr * CHK + jc] = f2bf(val);
      if (val != 0.0f) atomicAdd(&rowsum[tr], val);
    }
  }
  __syncthreads();

  // num = Pm x V + Qs x S_start  (accumulated in one f32 tile)
#pragma unroll
  for (int i = 0; i < 4; i++) {
    int tt = wv * 4 + i;
    int t0 = (tt >> 2) << 4, e0 = (tt & 3) << 4;
    v8f acc = (v8f){0, 0, 0, 0, 0, 0, 0, 0};
#pragma unroll
    for (int kk = 0; kk < CHK; kk += 32) {
      v16bf af = load_a_frag(Pm, CHK, t0, kk, lane);
      v16bf bfr = load_b_frag(VbT, CHK, e0, kk, lane);
      acc = __builtin_amdgcn_wmma_f32_16x16x32_bf16(false, af, false, bfr,
                                                    (short)0, acc, false, false);
    }
#pragma unroll
    for (int kk = 0; kk < HD; kk += 32) {
      v16bf af = load_a_frag(Qs, HD, t0, kk, lane);
      v16bf bfr = load_b_frag(SsT, HD, e0, kk, lane);
      acc = __builtin_amdgcn_wmma_f32_16x16x32_bf16(false, af, false, bfr,
                                                    (short)0, acc, false, false);
    }
#pragma unroll
    for (int r = 0; r < 8; r++) {
      int tr = t0 + half * 8 + r;
      int ec = e0 + lr;
      float den = aArr[tr] * pqz[tr] + rowsum[tr] + 1e-6f;
      int m = (c * CHK + tr) * BATCH + b;
      AO[(size_t)m * HID + h * HD + ec] = f2bf(acc[r] / den);
    }
  }
}

// ---------------------------------------------------------------------------
extern "C" void kernel_launch(void* const* d_in, const int* in_sizes, int n_in,
                              void* d_out, int out_size, void* d_ws, size_t ws_size,
                              hipStream_t stream) {
  (void)in_sizes; (void)n_in; (void)out_size; (void)ws_size;
  const float* hidden = (const float*)d_in[0];
  const float* Wq = (const float*)d_in[1];
  const float* bq = (const float*)d_in[2];
  const float* Wk = (const float*)d_in[3];
  const float* bk = (const float*)d_in[4];
  const float* Wv = (const float*)d_in[5];
  const float* bv = (const float*)d_in[6];
  const float* Wf = (const float*)d_in[7];
  const float* bfb = (const float*)d_in[8];
  const float* Wo = (const float*)d_in[9];
  const float* bo = (const float*)d_in[10];

  float* out0 = (float*)d_out;
  float* zeros = out0 + (size_t)MROWS * HID;
  float* fout = zeros + (size_t)MROWS * HID;  // [S,B,NH] fp32, 3rd output

  char* ws = (char*)d_ws;
  size_t off = 0;
  auto alloc = [&](size_t bytes) -> void* {
    void* p = ws + off;
    off += (bytes + 255) & ~(size_t)255;
    return p;
  };
  u16t* Xb   = (u16t*)alloc((size_t)MROWS * HID * 2);
  u16t* Wqb  = (u16t*)alloc((size_t)HID * HID * 2);
  u16t* Wkb  = (u16t*)alloc((size_t)HID * HID * 2);
  u16t* Wvb  = (u16t*)alloc((size_t)HID * HID * 2);
  u16t* Wob  = (u16t*)alloc((size_t)HID * HID * 2);
  u16t* Wfb  = (u16t*)alloc((size_t)NH * HID * 2);
  u16t* phiq = (u16t*)alloc((size_t)MROWS * HID * 2);
  u16t* phik = (u16t*)alloc((size_t)MROWS * HID * 2);
  u16t* vbuf = (u16t*)alloc((size_t)MROWS * HID * 2);
  float* Ubuf  = (float*)alloc((size_t)NCHK * BH * HD * HD * 4);
  float* zubuf = (float*)alloc((size_t)NCHK * BH * HD * 4);
  float* aCbuf = (float*)alloc((size_t)NCHK * BH * 4);
  float* Sck   = (float*)alloc((size_t)NCHK * BH * HD * HD * 4);
  float* zck   = (float*)alloc((size_t)NCHK * BH * HD * 4);
  u16t* AO   = (u16t*)alloc((size_t)MROWS * HID * 2);

  // 1) precision converts + zero second output
  convert_f32_bf16<<<2048, 256, 0, stream>>>(hidden, Xb, MROWS * HID);
  convert_f32_bf16<<<1024, 256, 0, stream>>>(Wq, Wqb, HID * HID);
  convert_f32_bf16<<<1024, 256, 0, stream>>>(Wk, Wkb, HID * HID);
  convert_f32_bf16<<<1024, 256, 0, stream>>>(Wv, Wvb, HID * HID);
  convert_f32_bf16<<<1024, 256, 0, stream>>>(Wo, Wob, HID * HID);
  convert_f32_bf16<<<64, 256, 0, stream>>>(Wf, Wfb, NH * HID);
  fill_zero_f32<<<2048, 256, 0, stream>>>(zeros, MROWS * HID);

  // 2) projections (WMMA): 64x32 strip per wave, 8 waves/block
  int blkQKV = (MROWS / 64) * (HID / 32) / 8;  // 512
  gemm_bf16_kernel<2><<<blkQKV, 256, 0, stream>>>(Xb, Wqb, bq, phiq, MROWS, HID, HID, 0);
  gemm_bf16_kernel<2><<<blkQKV, 256, 0, stream>>>(Xb, Wkb, bk, phik, MROWS, HID, HID, 0);
  gemm_bf16_kernel<2><<<blkQKV, 256, 0, stream>>>(Xb, Wvb, bv, vbuf, MROWS, HID, HID, 1);
  int blkF = (MROWS / 64) * (NH / 16) / 8;  // 16 (skinny: 64x16 per wave)
  gemm_bf16_kernel<1><<<blkF, 256, 0, stream>>>(Xb, Wfb, bfb, fout, MROWS, NH, HID, 3);

  // 3) chunked recurrence
  chunk_local_kernel<<<NCHK * BH, 128, 0, stream>>>(phik, vbuf, fout, Ubuf, zubuf, aCbuf);
  state_scan_kernel<<<BH, 256, 0, stream>>>(Ubuf, zubuf, aCbuf, Sck, zck);
  chunk_out_kernel<<<NCHK * BH, 128, 0, stream>>>(phiq, phik, vbuf, fout, Sck, zck, AO);

  // 4) output projection -> first output (fp32)
  gemm_bf16_kernel<2><<<blkQKV, 256, 0, stream>>>(AO, Wob, bo, out0, MROWS, HID, HID, 2);
}
